// MixtureOfExperts_82643760710107
// MI455X (gfx1250) — compile-verified, matrix operand order checked by reading, not run on hardware
//
#include <hip/hip_runtime.h>
#include <hip/hip_bf16.h>

#define NUM_EXPERTS 8
#define HIDDEN 1024
#define INTER 4096
#define NTOK 8192           // B*S = 4*2048
#define TILE_M 16

// LDS row strides (16-bit elements), padded, multiples of 8 for 16B alignment.
#define XS_STRIDE 1032      // 16 * 2064 B = 33,024 B
#define HS_STRIDE 4104      // 16 * 8208 B = 131,328 B
#define SMEM_BYTES (TILE_M*XS_STRIDE*2 + TILE_M*HS_STRIDE*2 + TILE_M*4 + TILE_M*4)

// Converted-weight tiling: a (32k x 16n) tile = 32 lanes * 16 bf16 = 1 KB,
// laid out so lane L's WMMA B-fragment is bytes [L*32, L*32+32).
#define KT1 (HIDDEN/32)     // 32  k-tiles per expert, GEMM1
#define FT1 (INTER/16)      // 256 f-tiles, GEMM1
#define KT2 (INTER/32)      // 128 k-tiles per expert, GEMM2
#define FT2 (HIDDEN/16)     // 64  d-tiles, GEMM2

typedef __attribute__((ext_vector_type(16))) __bf16        v16bf;
typedef __attribute__((ext_vector_type(8)))  float         v8f;
typedef __attribute__((ext_vector_type(4)))  unsigned int  v4u;

struct Frag32 { v4u lo, hi; };   // 32 B -> v16bf

__device__ __forceinline__ unsigned short f2bf(float f) {
    unsigned int u = __builtin_bit_cast(unsigned int, f);
    u += 0x7FFFu + ((u >> 16) & 1u);          // round-to-nearest-even
    return (unsigned short)(u >> 16);
}
__device__ __forceinline__ unsigned int f2bf2(float a, float b) {
    return (unsigned int)f2bf(a) | ((unsigned int)f2bf(b) << 16);
}
__device__ __forceinline__ v16bf frag_lds(const unsigned short* p) {
    Frag32 f; f.lo = *(const v4u*)p; f.hi = *(const v4u*)(p + 16);
    return __builtin_bit_cast(v16bf, f);
}
__device__ __forceinline__ v16bf frag_glb(const unsigned int* p) {
    Frag32 f; f.lo = *(const v4u*)p; f.hi = *(const v4u*)(p + 4);
    return __builtin_bit_cast(v16bf, f);
}
__device__ __forceinline__ v8f wmma_bf16(v16bf a, v16bf b, v8f c) {
    return __builtin_amdgcn_wmma_f32_16x16x32_bf16(false, a, false, b,
                                                   (short)0, c, false, false);
}

// ---------------------------------------------------------------- K0: init
__global__ void moe_zero(float* out, int n, int* cnts, float* psum) {
    int i = blockIdx.x * blockDim.x + threadIdx.x;
    for (; i < n; i += gridDim.x * blockDim.x) out[i] = 0.0f;
    if (blockIdx.x == 0 && threadIdx.x < NUM_EXPERTS) {
        cnts[threadIdx.x] = 0;
        psum[threadIdx.x] = 0.0f;
    }
}

// ----------------------------------------- K0b: fp32 -> bf16 fragment layout
// src: [ktiles*32][N] fp32 row-major. dst: u32[ktiles][ftiles][256] where
// u32 #t of a tile packs elements (j=2*jp, 2*jp+1) for lane = t>>3, jp = t&7.
__global__ void __launch_bounds__(256) conv_weights(
    const float* __restrict__ src, unsigned int* __restrict__ dst,
    int ktiles, int ftiles, int N)
{
    const int total = ktiles * ftiles * 256;
    for (int g = blockIdx.x * blockDim.x + threadIdx.x; g < total;
         g += gridDim.x * blockDim.x) {
        const int t    = g & 255;
        const int tt   = g >> 8;
        const int ft   = tt % ftiles;
        const int kt   = tt / ftiles;
        const int lane = t >> 3;
        const int jp   = t & 7;
        // fragment element j -> K index: khalf + (j<8 ? j : 16 + j-8)
        const int k   = ((lane >> 4) << 3) + ((jp < 4) ? (2 * jp) : (16 + 2 * (jp - 4)));
        const int row = kt * 32 + k;
        const int f   = ft * 16 + (lane & 15);
        const float* s = src + (size_t)row * N + f;
        dst[g] = f2bf2(s[0], s[(size_t)N]);
    }
}

// ------------------------------------------------- K1: router + bf16 convert
__global__ void __launch_bounds__(256) moe_router(
    const float* __restrict__ X, const float* __restrict__ Wr,
    unsigned short* __restrict__ Xbf, int* __restrict__ cnts,
    float* __restrict__ psum, int* __restrict__ tlist, float* __restrict__ wlist)
{
    const int wave = threadIdx.x >> 5;
    const int lane = threadIdx.x & 31;
    const int t    = blockIdx.x * 8 + wave;

    float acc[NUM_EXPERTS] = {};
    const float* xr = X + (size_t)t * HIDDEN;
#pragma unroll 1
    for (int d = lane; d < HIDDEN; d += 32) {
        float x = xr[d];
        Xbf[(size_t)t * HIDDEN + d] = f2bf(x);
        const float* wr = Wr + d * NUM_EXPERTS;
#pragma unroll
        for (int e = 0; e < NUM_EXPERTS; ++e) acc[e] += x * wr[e];
    }
#pragma unroll
    for (int off = 16; off > 0; off >>= 1)
#pragma unroll
        for (int e = 0; e < NUM_EXPERTS; ++e)
            acc[e] += __shfl_down(acc[e], off, 32);

    if (lane == 0) {
        float mx = acc[0];
#pragma unroll
        for (int e = 1; e < NUM_EXPERTS; ++e) mx = fmaxf(mx, acc[e]);
        float p[NUM_EXPERTS]; float s = 0.0f;
#pragma unroll
        for (int e = 0; e < NUM_EXPERTS; ++e) { p[e] = __expf(acc[e] - mx); s += p[e]; }
        const float inv = __builtin_amdgcn_rcpf(s);

        int i1 = 0;
#pragma unroll
        for (int e = 1; e < NUM_EXPERTS; ++e) if (p[e] > p[i1]) i1 = e;
        int i2 = (i1 == 0) ? 1 : 0;
#pragma unroll
        for (int e = 0; e < NUM_EXPERTS; ++e) if (e != i1 && p[e] > p[i2]) i2 = e;

        const float wn = __builtin_amdgcn_rcpf(p[i1] + p[i2]);
        int s1 = atomicAdd(&cnts[i1], 1);
        tlist[i1 * NTOK + s1] = t; wlist[i1 * NTOK + s1] = p[i1] * wn;
        int s2 = atomicAdd(&cnts[i2], 1);
        tlist[i2 * NTOK + s2] = t; wlist[i2 * NTOK + s2] = p[i2] * wn;
#pragma unroll
        for (int e = 0; e < NUM_EXPERTS; ++e) atomicAdd(&psum[e], p[e] * inv);
    }
}

// ---------------------------------------------------------------- K3: loss
__global__ void moe_loss(const int* cnts, const float* psum, float* out) {
    if (threadIdx.x == 0 && blockIdx.x == 0) {
        const float invT = 1.0f / (float)NTOK;
        float loss = 0.0f;
#pragma unroll
        for (int e = 0; e < NUM_EXPERTS; ++e)
            loss += ((float)cnts[e] * invT) * (psum[e] * invT);
        out[(size_t)NTOK * HIDDEN] = loss * (float)NUM_EXPERTS;
    }
}

// ------------------------------------------- K2: fused gathered expert FFN
// grid (NTOK/16, NUM_EXPERTS), 256 threads = 8 wave32s. Intermediate
// [16x4096] bf16 tile lives in LDS (CDNA5 WGP: 320 KB).
__global__ void __launch_bounds__(256) moe_ffn(
    const unsigned short* __restrict__ Xbf,
    const unsigned int* __restrict__ WgT, const unsigned int* __restrict__ WuT,
    const unsigned int* __restrict__ WdT,
    const int* __restrict__ cnts, const int* __restrict__ tlist,
    const float* __restrict__ wlist, float* __restrict__ out)
{
    extern __shared__ unsigned char smemRaw[];
    unsigned short* Xs   = (unsigned short*)smemRaw;           // [16][XS_STRIDE]
    unsigned short* Hs   = Xs + TILE_M * XS_STRIDE;            // [16][HS_STRIDE]
    int*            sTid = (int*)(Hs + TILE_M * HS_STRIDE);
    float*          sW   = (float*)(sTid + TILE_M);

    const int e   = blockIdx.y;
    const int cnt = cnts[e];
    const int m0  = blockIdx.x * TILE_M;
    if (m0 >= cnt) return;
    const int rows = min(TILE_M, cnt - m0);

    if (threadIdx.x < TILE_M) {
        int r = threadIdx.x;
        if (r < rows) {
            sTid[r] = tlist[e * NTOK + m0 + r];
            sW[r]   = wlist[e * NTOK + m0 + r];
        } else { sTid[r] = 0; sW[r] = 0.0f; }
    }
    __syncthreads();

    {   // Stage 16 gathered bf16 rows of X into LDS.
        const int r = threadIdx.x >> 4;
        const int c = threadIdx.x & 15;
        const v4u* src = (const v4u*)(Xbf + (size_t)sTid[r] * HIDDEN);
        v4u* dst = (v4u*)(Xs + r * XS_STRIDE);
#pragma unroll
        for (int j = 0; j < 8; ++j) {
            v4u v = {0u, 0u, 0u, 0u};
            if (r < rows) v = src[c + 16 * j];
            dst[c + 16 * j] = v;
        }
    }
    __syncthreads();

    const int waveId = threadIdx.x >> 5;
    const int lane   = threadIdx.x & 31;
    const int nlo    = lane & 15;
    const int khalf  = (lane >> 4) * 8;
    const int lofs   = lane * 8;                 // u32 offset of this lane's fragment

    // -------- GEMM1: H = silu(X*Wg) .* (X*Wu); two f-tiles per wave-iter --------
    // k-loop unrolled 2x with two independent register buffers (A/B sets):
    // no cross-set assignments => no register copies; loads overlap WMMAs.
    const size_t kstep1 = (size_t)FT1 * 256;     // u32 per k-tile advance
#pragma unroll 1
    for (int cp = waveId; cp < INTER / 32; cp += 8) {
        const int ft0 = cp * 2, ft1 = cp * 2 + 1;
        const unsigned int* pg = WgT + ((size_t)e * KT1 * FT1 + ft0) * 256 + lofs;
        const unsigned int* pu = WuT + ((size_t)e * KT1 * FT1 + ft0) * 256 + lofs;

        v8f aG0 = {}, aU0 = {}, aG1 = {}, aU1 = {};
        v16bf gA0 = frag_glb(pg), gA1 = frag_glb(pg + 256);
        v16bf uA0 = frag_glb(pu), uA1 = frag_glb(pu + 256);
#pragma unroll 1
        for (int kk = 0; kk < HIDDEN; kk += 64) {
            // B set: k-tile kk+32 (always in range: k-tile count is even)
            v16bf gB0 = frag_glb(pg + kstep1), gB1 = frag_glb(pg + kstep1 + 256);
            v16bf uB0 = frag_glb(pu + kstep1), uB1 = frag_glb(pu + kstep1 + 256);
            v16bf a0 = frag_lds(Xs + nlo * XS_STRIDE + kk + khalf);
            aG0 = wmma_bf16(a0, gA0, aG0);
            aU0 = wmma_bf16(a0, uA0, aU0);
            aG1 = wmma_bf16(a0, gA1, aG1);
            aU1 = wmma_bf16(a0, uA1, aU1);
            if (kk + 64 < HIDDEN) {             // A set: k-tile kk+64
                pg += 2 * kstep1; pu += 2 * kstep1;
                gA0 = frag_glb(pg); gA1 = frag_glb(pg + 256);
                uA0 = frag_glb(pu); uA1 = frag_glb(pu + 256);
                __builtin_prefetch(pg + 2 * kstep1, 0, 3);
                __builtin_prefetch(pu + 2 * kstep1, 0, 3);
            }
            v16bf a1 = frag_lds(Xs + nlo * XS_STRIDE + kk + 32 + khalf);
            aG0 = wmma_bf16(a1, gB0, aG0);
            aU0 = wmma_bf16(a1, uB0, aU0);
            aG1 = wmma_bf16(a1, gB1, aG1);
            aU1 = wmma_bf16(a1, uB1, aU1);
        }
        // C layout: lane holds N = lane&15, M = 8*(lane>>4) + v
#pragma unroll
        for (int v = 0; v < 8; ++v) {
            const int m = khalf + v;
            float g0 = aG0[v];
            float h0 = g0 * __builtin_amdgcn_rcpf(1.0f + __expf(-g0)) * aU0[v];
            Hs[m * HS_STRIDE + ft0 * 16 + nlo] = f2bf(h0);
            float g1 = aG1[v];
            float h1 = g1 * __builtin_amdgcn_rcpf(1.0f + __expf(-g1)) * aU1[v];
            Hs[m * HS_STRIDE + ft1 * 16 + nlo] = f2bf(h1);
        }
    }
    __syncthreads();

    // -------- GEMM2: out += w_tok * (H * Wd); two d-tiles per wave-iter --------
    const size_t kstep2 = (size_t)FT2 * 256;
#pragma unroll 1
    for (int cp = waveId; cp < HIDDEN / 32; cp += 8) {
        const int dt0 = cp * 2, dt1 = cp * 2 + 1;
        const unsigned int* pd = WdT + ((size_t)e * KT2 * FT2 + dt0) * 256 + lofs;

        v8f a0 = {}, a1 = {};
        v16bf dA0 = frag_glb(pd), dA1 = frag_glb(pd + 256);
#pragma unroll 1
        for (int kk = 0; kk < INTER; kk += 64) {
            v16bf dB0 = frag_glb(pd + kstep2), dB1 = frag_glb(pd + kstep2 + 256);
            v16bf x0 = frag_lds(Hs + nlo * HS_STRIDE + kk + khalf);
            a0 = wmma_bf16(x0, dA0, a0);
            a1 = wmma_bf16(x0, dA1, a1);
            if (kk + 64 < INTER) {
                pd += 2 * kstep2;
                dA0 = frag_glb(pd); dA1 = frag_glb(pd + 256);
                __builtin_prefetch(pd + 2 * kstep2, 0, 3);
            }
            v16bf x1 = frag_lds(Hs + nlo * HS_STRIDE + kk + 32 + khalf);
            a0 = wmma_bf16(x1, dB0, a0);
            a1 = wmma_bf16(x1, dB1, a1);
        }
#pragma unroll
        for (int v = 0; v < 8; ++v) {
            const int m = khalf + v;
            if (m < rows) {
                const float w = sW[m];
                float* orow = out + (size_t)sTid[m] * HIDDEN;
                atomicAdd(orow + dt0 * 16 + nlo, a0[v] * w);
                atomicAdd(orow + dt1 * 16 + nlo, a1[v] * w);
            }
        }
    }
}

// ---------------------------------------------------------------- launcher
extern "C" void kernel_launch(void* const* d_in, const int* in_sizes, int n_in,
                              void* d_out, int out_size, void* d_ws, size_t ws_size,
                              hipStream_t stream) {
    const float* X  = (const float*)d_in[0];
    const float* Wr = (const float*)d_in[1];
    const float* Wg = (const float*)d_in[2];
    const float* Wu = (const float*)d_in[3];
    const float* Wd = (const float*)d_in[4];
    float* out = (float*)d_out;

    // Workspace layout
    char* ws = (char*)d_ws;
    const size_t WSZ = (size_t)NUM_EXPERTS * HIDDEN * INTER * 2; // 64 MB bf16
    int*            cnts  = (int*)ws;
    float*          psum  = (float*)(ws + 64);
    int*            tlist = (int*)(ws + 128);
    float*          wlist = (float*)(ws + 128 + (size_t)NUM_EXPERTS * NTOK * 4);
    unsigned short* Xbf   = (unsigned short*)(ws + 128 + 2ull * NUM_EXPERTS * NTOK * 4);
    unsigned int*   WgT   = (unsigned int*)((char*)Xbf + (size_t)NTOK * HIDDEN * 2);
    unsigned int*   WuT   = (unsigned int*)((char*)WgT + WSZ);
    unsigned int*   WdT   = (unsigned int*)((char*)WuT + WSZ);

    const int out_elems = NTOK * HIDDEN + 1;
    moe_zero<<<2048, 256, 0, stream>>>(out, out_elems, cnts, psum);

    // Weight conversion to WMMA fragment layout (bf16).
    conv_weights<<<16384, 256, 0, stream>>>(Wg, WgT, NUM_EXPERTS * KT1, FT1, INTER);
    conv_weights<<<16384, 256, 0, stream>>>(Wu, WuT, NUM_EXPERTS * KT1, FT1, INTER);
    conv_weights<<<16384, 256, 0, stream>>>(Wd, WdT, NUM_EXPERTS * KT2, FT2, HIDDEN);

    moe_router<<<NTOK / 8, 256, 0, stream>>>(X, Wr, Xbf, cnts, psum, tlist, wlist);
    moe_loss<<<1, 32, 0, stream>>>(cnts, psum, out);

    dim3 grid(NTOK / TILE_M, NUM_EXPERTS);
    moe_ffn<<<grid, 256, SMEM_BYTES, stream>>>(Xbf, WgT, WuT, WdT,
                                               cnts, tlist, wlist, out);
    (void)in_sizes; (void)n_in; (void)out_size; (void)ws_size;
}
